// TraverseSubGNN_21388937134520
// MI455X (gfx1250) — matrix operand order repaired and checked by reading, not compiled, not run on hardware
//
#include <hip/hip_runtime.h>
#include <hip/hip_bf16.h>

#define HEADS   3
#define EMBD    192
#define KDIM    64
#define DIN     128
#define MAXLAYER 4
#define SUBG    500
#define NEG_ATT 0.2f
#define NEG_ACT 0.1f
#define BN_EPS  1e-5f

typedef __attribute__((ext_vector_type(16))) __bf16 v16bf;
typedef __attribute__((ext_vector_type(8)))  float  v8f;
typedef __attribute__((ext_vector_type(4)))  unsigned int v4u32;
typedef __attribute__((ext_vector_type(8)))  int  v8i32;
typedef __attribute__((ext_vector_type(4)))  int  v4i32;

__device__ __forceinline__ float leakyf(float x, float s) { return x >= 0.f ? x : s * x; }

__device__ __forceinline__ void atomicMaxF(float* addr, float val) {
    unsigned int* ua = (unsigned int*)addr;
    unsigned int cur = __float_as_uint(*addr);
    while (__uint_as_float(cur) < val) {
        unsigned int prev = atomicCAS(ua, cur, __float_as_uint(val));
        if (prev == cur) break;
        cur = prev;
    }
}

// ---------------- init ----------------
__global__ __launch_bounds__(256) void init_kernel(float* __restrict__ aggr, long long nA,
                                                   float* __restrict__ amax, float* __restrict__ denom,
                                                   long long nS, float* __restrict__ chan) {
    long long i = (long long)blockIdx.x * blockDim.x + threadIdx.x;
    if (i < nA) aggr[i] = 0.f;
    if (i < nS) { amax[i] = -3.0e38f; denom[i] = 0.f; }
    if (i < 2 * EMBD) chan[i] = 0.f;
}

// ---------------- GEMM: x[N,192] = subx[N,128] @ W[192,128]^T  (TDM A-stage + bf16 WMMA) ----------------
__global__ __launch_bounds__(256) void gemm_x_kernel(const float* __restrict__ subx,
                                                     const float* __restrict__ W,
                                                     float* __restrict__ x, int N) {
    __shared__ __bf16 sW[EMBD * DIN];   // 48 KB : whole weight matrix, bf16
    __shared__ float  sAf[128 * DIN];   // 64 KB : 128 input rows, f32 (TDM target)
    const int tid = threadIdx.x;
    const int rowbase = blockIdx.x * 128;
    const int rows = (N - rowbase < 128) ? (N - rowbase) : 128;

    // W -> LDS (bf16) cooperatively
    for (int i = tid; i < EMBD * DIN; i += 256) sW[i] = (__bf16)W[i];

    // A tile (rows x 128 f32) -> LDS via Tensor Data Mover, issued by wave 0 only
    if (tid < 32) {
        unsigned long long gaddr = (unsigned long long)(const void*)(subx + (long long)rowbase * DIN);
        unsigned int lds_off = (unsigned int)(unsigned long long)(uintptr_t)&sAf[0];
        unsigned int tdim1 = (unsigned int)(N - rowbase);   // remaining tensor rows from tile start

        v4u32 g0;
        g0[0] = 1u;                                           // count=1 (valid user descriptor)
        g0[1] = lds_off;                                      // lds_addr (bytes)
        g0[2] = (unsigned int)(gaddr & 0xffffffffu);          // global_addr[31:0]
        g0[3] = (unsigned int)((gaddr >> 32) & 0x01ffffffu)   // global_addr[56:32]
                | (2u << 30);                                 // type = 2 ("image")

        v8i32 g1 = {};
        g1[0] = (int)(2u << 16);                              // data_size = 2 -> 4 bytes
        g1[1] = (int)((unsigned)DIN << 16);                   // tensor_dim0[15:0] = 128
        g1[2] = (int)((tdim1 & 0xffffu) << 16);               // tensor_dim1[15:0]
        g1[3] = (int)((tdim1 >> 16) & 0xffffu)                // tensor_dim1[31:16]
                | (DIN << 16);                                // tile_dim0 = 128
        g1[4] = (int)(unsigned)rows;                          // tile_dim1 = rows (tile_dim2=0)
        g1[5] = DIN;                                          // tensor_dim0_stride = 128
        g1[6] = 0;
        g1[7] = 0;

        v4i32 gz4 = {};
        v8i32 gz8 = {};
        __builtin_amdgcn_tensor_load_to_lds(g0, g1, gz4, gz4, gz8, 0);
        __builtin_amdgcn_s_wait_tensorcnt(0);
    }
    __syncthreads();

    const int wave = tid >> 5;
    const int lane = tid & 31;
    const int row  = lane & 15;          // M row (A) / N col (B,C)
    const int hi   = (lane & 16) ? 1 : 0;

    // A fragments (ISA 16-bit A 16x32 layout): lanes 0-15 hold K {0..7,16..23}, lanes 16-31 hold K {8..15,24..31}
    v16bf a4[4];
#pragma unroll
    for (int kt = 0; kt < 4; ++kt) {
        const int kb = kt * 32 + hi * 8;
#pragma unroll
        for (int i = 0; i < 8; ++i) {
            int koff = (i < 4) ? (2 * i) : (16 + 2 * (i - 4));
            a4[kt][2 * i]     = (__bf16)sAf[(wave * 16 + row) * DIN + kb + koff];
            a4[kt][2 * i + 1] = (__bf16)sAf[(wave * 16 + row) * DIN + kb + koff + 1];
        }
    }

    const int gr = rowbase + wave * 16 + hi * 8;   // this lane's first output row
    const bool full = (rowbase + 128 <= N);

#pragma unroll
    for (int nt = 0; nt < EMBD / 16; ++nt) {
        v8f c = {};
#pragma unroll
        for (int kt = 0; kt < 4; ++kt) {
            // B fragment (16-bit B 32x16): lane n -> column n&15; lanes 0-15 K {0..7,16..23}, lanes 16-31 K {8..15,24..31}
            v16bf b;
            const int kb2 = kt * 32 + hi * 8;
#pragma unroll
            for (int v = 0; v < 8; ++v) {
                int koff2 = (v < 4) ? (2 * v) : (16 + 2 * (v - 4));
                b[2 * v]     = sW[(nt * 16 + row) * DIN + kb2 + koff2];
                b[2 * v + 1] = sW[(nt * 16 + row) * DIN + kb2 + koff2 + 1];
            }
            c = __builtin_amdgcn_wmma_f32_16x16x32_bf16(false, a4[kt], false, b,
                                                        (short)0, c, false, false);
        }
        // C layout: VGPR r, lane L: row = r + (L>=16 ? 8:0), col = L&15
        if (full) {
            float* op = x + (long long)gr * EMBD + nt * 16 + row;
#pragma unroll
            for (int r = 0; r < 8; ++r) op[(long long)r * EMBD] = c[r];
        } else {
#pragma unroll
            for (int r = 0; r < 8; ++r) {
                if (gr + r < N) x[(long long)(gr + r) * EMBD + nt * 16 + row] = c[r];
            }
        }
    }
}

// ---------------- per-node attention scores ----------------
__global__ __launch_bounds__(256) void scores_kernel(const float* __restrict__ x,
                                                     const float* __restrict__ att,
                                                     float* __restrict__ sdst,
                                                     float* __restrict__ ssrc, int N) {
    int idx = blockIdx.x * blockDim.x + threadIdx.x;
    if (idx >= N * HEADS) return;
    int n = idx / HEADS, h = idx % HEADS;
    const float* xp = x + (long long)n * EMBD + h * KDIM;
    const float* ai = att + h * 2 * KDIM;       // x_i half
    const float* aj = ai + KDIM;                // x_j half
    if ((n & 63) == 0) __builtin_prefetch(xp + 64 * EMBD, 0, 1);
    float si = 0.f, sj = 0.f;
#pragma unroll 8
    for (int j = 0; j < KDIM; ++j) { float v = xp[j]; si += v * ai[j]; sj += v * aj[j]; }
    sdst[idx] = si;
    ssrc[idx] = sj;
}

// ---------------- edge logits + segment max (grouped by src) ----------------
__global__ __launch_bounds__(256) void logits_kernel(const float* __restrict__ sdst,
                                                     const float* __restrict__ ssrc,
                                                     const int* __restrict__ ei,
                                                     float* __restrict__ alpha,
                                                     float* __restrict__ amax, int E) {
    int idx = blockIdx.x * blockDim.x + threadIdx.x;
    if (idx >= E * HEADS) return;
    int e = idx / HEADS, h = idx % HEADS;
    int s = ei[e], d = ei[E + e];
    float a = leakyf(sdst[d * HEADS + h] + ssrc[s * HEADS + h], NEG_ATT);
    alpha[idx] = a;
    atomicMaxF(&amax[s * HEADS + h], a);
}

// ---------------- exp + segment sum ----------------
__global__ __launch_bounds__(256) void expsum_kernel(const float* __restrict__ alpha,
                                                     const int* __restrict__ ei,
                                                     const float* __restrict__ amax,
                                                     float* __restrict__ denom, int E) {
    int idx = blockIdx.x * blockDim.x + threadIdx.x;
    if (idx >= E * HEADS) return;
    int e = idx / HEADS, h = idx % HEADS;
    int s = ei[e];
    float ex = __expf(alpha[idx] - amax[s * HEADS + h]);
    atomicAdd(&denom[s * HEADS + h], ex);
}

// ---------------- masked message scatter-add: 64 threads per edge ----------------
__global__ __launch_bounds__(256) void message_kernel(const float* __restrict__ x,
                                                      const int* __restrict__ ei,
                                                      const int* __restrict__ mask,
                                                      const float* __restrict__ alpha,
                                                      const float* __restrict__ amax,
                                                      const float* __restrict__ denom,
                                                      float* __restrict__ aggr, int E) {
    long long gid = (long long)blockIdx.x * blockDim.x + threadIdx.x;
    int e = (int)(gid >> 6);
    int t = (int)(gid & 63);
    if (e >= E) return;
    if (mask[e] == 0) return;
    int s = ei[e], d = ei[E + e];
    const float* xs = x + (long long)s * EMBD;
    float* ad = aggr + (long long)d * EMBD;
    if (t == 0) __builtin_prefetch(xs, 0, 1);   // global_prefetch_b8 (x is L2-resident)
#pragma unroll
    for (int h = 0; h < HEADS; ++h) {
        float coef = __expf(alpha[e * HEADS + h] - amax[s * HEADS + h]) /
                     (denom[s * HEADS + h] + 1e-16f);
        int f = h * KDIM + t;
        atomicAdd(&ad[f], coef * xs[f]);
    }
}

// ---------------- bias + LeakyReLU(0.1) in-place + per-channel stats ----------------
__global__ __launch_bounds__(256) void act_stats_kernel(float* __restrict__ aggr,
                                                        const float* __restrict__ bias,
                                                        float* __restrict__ chan, int N) {
    __shared__ float lsum[EMBD];
    __shared__ float lsq[EMBD];
    int tid = threadIdx.x;
    for (int i = tid; i < EMBD; i += 256) { lsum[i] = 0.f; lsq[i] = 0.f; }
    __syncthreads();
    const long long total = (long long)N * EMBD;
    long long base = (long long)blockIdx.x * 256 * 96;
    for (int it = 0; it < 96; ++it) {
        long long idx = base + (long long)it * 256 + tid;
        if (idx < total) {
            int f = (int)(idx % EMBD);
            float h = leakyf(aggr[idx] + bias[f], NEG_ACT);
            aggr[idx] = h;
            atomicAdd(&lsum[f], h);
            atomicAdd(&lsq[f], h * h);
        }
    }
    __syncthreads();
    for (int i = tid; i < EMBD; i += 256) {
        atomicAdd(&chan[i], lsum[i]);
        atomicAdd(&chan[EMBD + i], lsq[i]);
    }
}

__global__ void finalize_kernel(const float* __restrict__ chan, float* __restrict__ mr, int N) {
    int f = threadIdx.x;
    if (f >= EMBD) return;
    float inv = 1.f / (float)N;
    float mean = chan[f] * inv;
    float var  = chan[EMBD + f] * inv - mean * mean;
    mr[f] = mean;
    mr[EMBD + f] = rsqrtf(var + BN_EPS);
}

// ---------------- layer-mean + batchnorm affine (commutes with mean) ----------------
__global__ __launch_bounds__(256) void output_kernel(const float* __restrict__ h,
                                                     const float* __restrict__ gamma,
                                                     const float* __restrict__ beta,
                                                     const float* __restrict__ mr,
                                                     float* __restrict__ out, int OUTN) {
    int idx = blockIdx.x * blockDim.x + threadIdx.x;
    if (idx >= OUTN * EMBD) return;
    int f = idx % EMBD;
    int o = idx / EMBD;
    int b = o / SUBG, s = o % SUBG;
    float acc = 0.f;
#pragma unroll
    for (int l = 0; l < MAXLAYER; ++l) {
        long long n = (long long)b * (MAXLAYER * SUBG) + (long long)l * SUBG + s;
        acc += h[n * EMBD + f];
    }
    acc *= (1.f / MAXLAYER);
    out[idx] = gamma[f] * (acc - mr[f]) * mr[EMBD + f] + beta[f];
}

extern "C" void kernel_launch(void* const* d_in, const int* in_sizes, int n_in,
                              void* d_out, int out_size, void* d_ws, size_t ws_size,
                              hipStream_t stream) {
    const float* subx  = (const float*)d_in[0];
    const int*   ei    = (const int*)d_in[1];
    const int*   mask  = (const int*)d_in[2];
    const float* W     = (const float*)d_in[3];
    const float* att   = (const float*)d_in[4];
    const float* bias  = (const float*)d_in[5];
    const float* gamma = (const float*)d_in[6];
    const float* beta  = (const float*)d_in[7];
    float* out = (float*)d_out;

    const int N = in_sizes[0] / DIN;
    const int E = in_sizes[1] / 2;

    float* ws    = (float*)d_ws;
    float* x     = ws;                                  // N*EMBD
    float* sdst  = x + (size_t)N * EMBD;                // N*HEADS
    float* ssrc  = sdst + (size_t)N * HEADS;            // N*HEADS
    float* amax  = ssrc + (size_t)N * HEADS;            // N*HEADS
    float* denom = amax + (size_t)N * HEADS;            // N*HEADS
    float* alpha = denom + (size_t)N * HEADS;           // E*HEADS
    float* aggr  = alpha + (size_t)E * HEADS;           // N*EMBD
    float* chan  = aggr + (size_t)N * EMBD;             // 2*EMBD
    float* mr    = chan + 2 * EMBD;                     // 2*EMBD

    const long long nAggr = (long long)N * EMBD;
    const long long nScore = (long long)N * HEADS;

    {
        long long blocks = (nAggr + 255) / 256;
        init_kernel<<<(int)blocks, 256, 0, stream>>>(aggr, nAggr, amax, denom, nScore, chan);
    }
    gemm_x_kernel<<<(N + 127) / 128, 256, 0, stream>>>(subx, W, x, N);
    scores_kernel<<<(N * HEADS + 255) / 256, 256, 0, stream>>>(x, att, sdst, ssrc, N);
    logits_kernel<<<(E * HEADS + 255) / 256, 256, 0, stream>>>(sdst, ssrc, ei, alpha, amax, E);
    expsum_kernel<<<(E * HEADS + 255) / 256, 256, 0, stream>>>(alpha, ei, amax, denom, E);
    {
        long long threads = (long long)E * 64;
        message_kernel<<<(int)((threads + 255) / 256), 256, 0, stream>>>(x, ei, mask, alpha, amax,
                                                                         denom, aggr, E);
    }
    {
        long long blocks = (nAggr + 24575) / 24576;
        act_stats_kernel<<<(int)blocks, 256, 0, stream>>>(aggr, bias, chan, N);
    }
    finalize_kernel<<<1, EMBD, 0, stream>>>(chan, mr, N);
    {
        int OUTN = out_size / EMBD;
        output_kernel<<<(out_size + 255) / 256, 256, 0, stream>>>(aggr, gamma, beta, mr, out, OUTN);
    }
}
